// VTDCLIP_81956565942962
// MI455X (gfx1250) — compile-verified, matrix operand order checked by reading, not run on hardware
//
#include <hip/hip_runtime.h>
#include <math.h>

// ---------------- types ----------------
typedef unsigned short u16;
typedef __attribute__((ext_vector_type(16))) __bf16 v16bf;
typedef __attribute__((ext_vector_type(8)))  float  v8f;
typedef __attribute__((ext_vector_type(8)))  u16    ushort8x;

union Frag { ushort8x h[2]; v16bf v; };

// float -> bf16 (round to nearest even)
__device__ __forceinline__ u16 f2bf(float f) {
    union { float f; unsigned u; } x; x.f = f;
    unsigned u = x.u;
    unsigned r = u + 0x7FFFu + ((u >> 16) & 1u);
    return (u16)(r >> 16);
}

// ---------------- problem constants ----------------
#define BATCH 1024
#define TFRM  16
#define DMODEL 512
#define NCLS  400
#define HID   2048
#define MROWS (BATCH * TFRM)   // 16384

// ============================================================
// Row L2-norm + bf16 conversion (one wave32 per row)
// ============================================================
__global__ void rownorm_bf16_kernel(const float* __restrict__ src,
                                    u16* __restrict__ dstb,
                                    float* __restrict__ norms,
                                    float* __restrict__ sq,
                                    int rows, int cols) {
    int row  = blockIdx.x * (blockDim.x >> 5) + (threadIdx.x >> 5);
    int lane = threadIdx.x & 31;
    if (row >= rows) return;
    const float* s = src + (size_t)row * cols;
    u16* d = dstb + (size_t)row * cols;
    float acc = 0.f;
    for (int k = lane; k < cols; k += 32) {
        float v = s[k];
        acc += v * v;
        d[k] = f2bf(v);
    }
    for (int o = 16; o; o >>= 1) acc += __shfl_xor(acc, o, 32);
    if (lane == 0) {
        if (norms) norms[row] = sqrtf(acc);
        if (sq)    sq[row]    = acc;
    }
}

// ============================================================
// Elementwise f32 -> bf16 (weights)
// ============================================================
__global__ void cvt_bf16_kernel(const float* __restrict__ s, u16* __restrict__ d, size_t n) {
    size_t i = (size_t)blockIdx.x * blockDim.x + threadIdx.x;
    size_t stride = (size_t)gridDim.x * blockDim.x;
    for (; i < n; i += stride) d[i] = f2bf(s[i]);
}

// ============================================================
// WMMA bf16 GEMM: C[M,N] = A[M,K] * B[N,K]^T (+bias, relu)
//  - Cf : f32 output (ld = N), optional
//  - Cb : bf16 mirror of output, optional (feeds next GEMM)
//  - Ct : transposed, scaled output: Ct[n*ldt + m] = C/(rnorm[m]*cnorm[n])
// Block = 128 threads (4 waves along M). Each wave owns a 32x32 output tile
// (2x2 WMMA accumulators): 4 WMMAs per 8 b128 loads -> 16 FLOP/byte fetched.
// Block tile = 128(M) x 32(N).
// ============================================================
__global__ void wmma_gemm_bf16(const u16* __restrict__ A, int lda,
                               const u16* __restrict__ B, int ldb,
                               int M, int N, int K,
                               const float* __restrict__ bias, int relu,
                               float* __restrict__ Cf, u16* __restrict__ Cb,
                               float* __restrict__ Ct, int ldt,
                               const float* __restrict__ rnorm,
                               const float* __restrict__ cnorm) {
    int lane = threadIdx.x & 31;
    int wave = threadIdx.x >> 5;
    int tm = blockIdx.x * 128 + wave * 32;  // wave-uniform
    int tn = blockIdx.y * 32;
    if (tm >= M || tn >= N) return;         // wave-uniform exit: EXEC stays full

    // fragment source rows (clamped for ragged edges; stores are guarded below)
    int mr0 = tm      + (lane & 15); if (mr0 >= M) mr0 = M - 1;
    int mr1 = tm + 16 + (lane & 15); if (mr1 >= M) mr1 = M - 1;
    int nr0 = tn      + (lane & 15); if (nr0 >= N) nr0 = N - 1;
    int nr1 = tn + 16 + (lane & 15); if (nr1 >= N) nr1 = N - 1;
    int koff = (lane & 16) ? 8 : 0;         // K half-split per half-wave
    const u16* arow0 = A + (size_t)mr0 * lda + koff;
    const u16* arow1 = A + (size_t)mr1 * lda + koff;
    const u16* brow0 = B + (size_t)nr0 * ldb + koff;
    const u16* brow1 = B + (size_t)nr1 * ldb + koff;

    v8f acc00 = {0.f,0.f,0.f,0.f,0.f,0.f,0.f,0.f};
    v8f acc01 = acc00, acc10 = acc00, acc11 = acc00;

    #pragma unroll 2
    for (int k = 0; k < K; k += 32) {
        Frag a0, a1, b0, b1;
        a0.h[0] = *(const ushort8x*)(arow0 + k);       // K = koff..koff+7
        a0.h[1] = *(const ushort8x*)(arow0 + k + 16);  // K = koff+16..koff+23
        a1.h[0] = *(const ushort8x*)(arow1 + k);
        a1.h[1] = *(const ushort8x*)(arow1 + k + 16);
        b0.h[0] = *(const ushort8x*)(brow0 + k);
        b0.h[1] = *(const ushort8x*)(brow0 + k + 16);
        b1.h[0] = *(const ushort8x*)(brow1 + k);
        b1.h[1] = *(const ushort8x*)(brow1 + k + 16);
        acc00 = __builtin_amdgcn_wmma_f32_16x16x32_bf16(false, a0.v, false, b0.v, (short)0, acc00, false, false);
        acc01 = __builtin_amdgcn_wmma_f32_16x16x32_bf16(false, a0.v, false, b1.v, (short)0, acc01, false, false);
        acc10 = __builtin_amdgcn_wmma_f32_16x16x32_bf16(false, a1.v, false, b0.v, (short)0, acc10, false, false);
        acc11 = __builtin_amdgcn_wmma_f32_16x16x32_bf16(false, a1.v, false, b1.v, (short)0, acc11, false, false);
    }

    // C/D layout per 16x16 tile: lanes 0-15 -> N=lane, M=base+r ;
    // lanes 16-31 -> N=lane-16, M=base+8+r
    int nlo   = lane & 15;
    int mhalf = (lane & 16) ? 8 : 0;
    #pragma unroll
    for (int q = 0; q < 4; ++q) {
        v8f* accp = (q == 0) ? &acc00 : (q == 1) ? &acc01 : (q == 2) ? &acc10 : &acc11;
        int tmb = tm + ((q >> 1) ? 16 : 0);
        int tnb = tn + ((q & 1) ? 16 : 0);
        int ncol = tnb + nlo;
        if (ncol >= N) continue;            // after all WMMA: divergence is safe
        float bv = bias ? bias[ncol] : 0.f;
        float cn = cnorm ? cnorm[ncol] : 1.f;
        int mbase = tmb + mhalf;
        #pragma unroll
        for (int r = 0; r < 8; ++r) {
            int m = mbase + r;
            if (m < M) {
                float v = (*accp)[r] + bv;
                if (relu) v = fmaxf(v, 0.f);
                if (Cf) Cf[(size_t)m * N + ncol] = v;
                if (Cb) Cb[(size_t)m * N + ncol] = f2bf(v);
                if (Ct) Ct[(size_t)ncol * (size_t)ldt + m] =
                            v / (cn * (rnorm ? rnorm[m] : 1.f));
            }
        }
    }
}

// ============================================================
// VQ argmin over classes: min_idx[row] = argmin_n (c2[n] - 2*S[row,n])
// one wave32 per row
// ============================================================
__global__ void argmin_kernel(const float* __restrict__ S,
                              const float* __restrict__ c2,
                              int* __restrict__ min_idx,
                              int rows, int ncls) {
    int row  = blockIdx.x * (blockDim.x >> 5) + (threadIdx.x >> 5);
    int lane = threadIdx.x & 31;
    if (row >= rows) return;
    const float* s = S + (size_t)row * ncls;
    float bv = INFINITY; int bi = 0x7fffffff;
    for (int n = lane; n < ncls; n += 32) {
        float v = c2[n] - 2.f * s[n];
        if (v < bv || (v == bv && n < bi)) { bv = v; bi = n; }
    }
    for (int o = 16; o; o >>= 1) {
        float ov = __shfl_xor(bv, o, 32);
        int   oi = __shfl_xor(bi, o, 32);
        if (ov < bv || (ov == bv && oi < bi)) { bv = ov; bi = oi; }
    }
    if (lane == 0) min_idx[row] = bi;
}

// ============================================================
// Per-batch top-1 class: video_cls[b,n] = sum_t [min_idx==n]*softmax_t(sims/.01)
// nonzero only for candidate classes {min_idx[b,t]}, all values > 0
// => argmax restricted to candidates (tie -> smaller n, matching jnp.argmax).
// one wave32 per batch; lane t<16 evaluates candidate min_idx[b,t].
// ============================================================
__global__ void topidx_kernel(const float* __restrict__ S,
                              const int* __restrict__ min_idx,
                              const float* __restrict__ nimg,
                              const float* __restrict__ ntxt,
                              int* __restrict__ top_idx,
                              int Bn, int ncls) {
    int b    = blockIdx.x * (blockDim.x >> 5) + (threadIdx.x >> 5);
    int lane = threadIdx.x & 31;
    if (b >= Bn) return;
    float score = -INFINITY; int nid = 0x7fffffff;
    if (lane < TFRM) {
        int n = min_idx[b * TFRM + lane];
        float inv_tn = 100.f / ntxt[n];   // includes 1/0.01 temperature
        float sv[TFRM];
        float smax = -INFINITY;
        #pragma unroll
        for (int tp = 0; tp < TFRM; ++tp) {
            int r = b * TFRM + tp;
            float sim = (S[(size_t)r * ncls + n] / nimg[r]) * inv_tn;
            sv[tp] = sim;
            smax = fmaxf(smax, sim);
        }
        float den = 0.f, num = 0.f;
        #pragma unroll
        for (int tp = 0; tp < TFRM; ++tp) {
            float e = __expf(sv[tp] - smax);
            den += e;
            if (min_idx[b * TFRM + tp] == n) num += e;
        }
        score = num / den;
        nid = n;
    }
    for (int o = 16; o; o >>= 1) {
        float os = __shfl_xor(score, o, 32);
        int   on = __shfl_xor(nid,   o, 32);
        if (os > score || (os == score && on < nid)) { score = os; nid = on; }
    }
    if (lane == 0) top_idx[b] = nid;
}

// ============================================================
// x = image_feats + vf[top_idx[b]] (broadcast over t); emit bf16 + row norm
// one wave32 per row (row = b*16 + t)
// ============================================================
__global__ void xbuild_kernel(const float* __restrict__ img,
                              const float* __restrict__ vf,
                              const int* __restrict__ top_idx,
                              u16* __restrict__ xb,
                              float* __restrict__ xnorm,
                              int rows, int cols) {
    int row  = blockIdx.x * (blockDim.x >> 5) + (threadIdx.x >> 5);
    int lane = threadIdx.x & 31;
    if (row >= rows) return;
    const float* vrow = vf + (size_t)top_idx[row / TFRM] * cols;
    const float* s = img + (size_t)row * cols;
    u16* d = xb + (size_t)row * cols;
    float acc = 0.f;
    for (int k = lane; k < cols; k += 32) {
        float v = s[k] + vrow[k];
        acc += v * v;
        d[k] = f2bf(v);
    }
    for (int o = 16; o; o >>= 1) acc += __shfl_xor(acc, o, 32);
    if (lane == 0) xnorm[row] = sqrtf(acc);
}

// ============================================================
// host launcher
// ============================================================
static inline size_t alignUp(size_t x, size_t a) { return (x + a - 1) & ~(a - 1); }
static inline unsigned ceilDiv(unsigned a, unsigned b) { return (a + b - 1) / b; }

extern "C" void kernel_launch(void* const* d_in, const int* in_sizes, int n_in,
                              void* d_out, int out_size, void* d_ws, size_t ws_size,
                              hipStream_t stream) {
    (void)in_sizes; (void)n_in; (void)out_size; (void)ws_size;
    const float* img   = (const float*)d_in[0];   // [1024,16,512]
    const float* txt   = (const float*)d_in[1];   // [400,512]
    // q_w(2)/q_b(3)/k_w(4)/k_b(5) are dead: tk==1 => softmax over one key == 1
    const float* v_w   = (const float*)d_in[6];
    const float* v_b   = (const float*)d_in[7];
    const float* o_w   = (const float*)d_in[8];
    const float* o_b   = (const float*)d_in[9];
    const float* fc1_w = (const float*)d_in[10];
    const float* fc1_b = (const float*)d_in[11];
    const float* fc2_w = (const float*)d_in[12];
    const float* fc2_b = (const float*)d_in[13];
    float* out = (float*)d_out;                   // [400,1024,16]

    // ---- workspace carve ----
    char* p = (char*)d_ws;
    size_t off = 0;
    auto carve = [&](size_t bytes) { void* r = p + off; off = alignUp(off + bytes, 256); return r; };
    u16*   imgb   = (u16*)  carve((size_t)MROWS * DMODEL * 2);
    u16*   txtb   = (u16*)  carve((size_t)NCLS  * DMODEL * 2);
    float* Sbuf   = (float*)carve((size_t)MROWS * NCLS * 4);
    float* nimg   = (float*)carve((size_t)MROWS * 4);
    float* ntxt   = (float*)carve((size_t)NCLS  * 4);
    float* c2     = (float*)carve((size_t)NCLS  * 4);
    int*   minidx = (int*)  carve((size_t)MROWS * 4);
    int*   topidx = (int*)  carve((size_t)BATCH * 4);
    u16*   vwb    = (u16*)  carve((size_t)DMODEL * DMODEL * 2);
    u16*   owb    = (u16*)  carve((size_t)DMODEL * DMODEL * 2);
    u16*   f1b    = (u16*)  carve((size_t)HID   * DMODEL * 2);
    u16*   f2b    = (u16*)  carve((size_t)DMODEL * HID   * 2);
    float* y1f    = (float*)carve((size_t)NCLS * DMODEL * 4);
    u16*   y1b    = (u16*)  carve((size_t)NCLS * DMODEL * 2);
    float* y2f    = (float*)carve((size_t)NCLS * DMODEL * 4);
    u16*   y2b    = (u16*)  carve((size_t)NCLS * DMODEL * 2);
    float* hf     = (float*)carve((size_t)NCLS * HID * 4);
    u16*   hb     = (u16*)  carve((size_t)NCLS * HID * 2);
    float* vff    = (float*)carve((size_t)NCLS * DMODEL * 4);
    u16*   xb     = (u16*)  carve((size_t)MROWS * DMODEL * 2);
    float* xnorm  = (float*)carve((size_t)MROWS * 4);

    const dim3 blk128(128), blk256(256);

    // 1) norms + bf16 conversions of activations
    rownorm_bf16_kernel<<<dim3(ceilDiv(MROWS, 8)), blk256, 0, stream>>>(img, imgb, nimg, nullptr, MROWS, DMODEL);
    rownorm_bf16_kernel<<<dim3(ceilDiv(NCLS, 8)),  blk256, 0, stream>>>(txt, txtb, ntxt, c2, NCLS, DMODEL);
    // weight conversions
    cvt_bf16_kernel<<<dim3(512),  blk256, 0, stream>>>(v_w,   vwb, (size_t)DMODEL * DMODEL);
    cvt_bf16_kernel<<<dim3(512),  blk256, 0, stream>>>(o_w,   owb, (size_t)DMODEL * DMODEL);
    cvt_bf16_kernel<<<dim3(1024), blk256, 0, stream>>>(fc1_w, f1b, (size_t)HID * DMODEL);
    cvt_bf16_kernel<<<dim3(1024), blk256, 0, stream>>>(fc2_w, f2b, (size_t)DMODEL * HID);

    // 2) S = image @ text^T  [16384 x 400], K=512  (raw dots; serves sims AND VQ)
    wmma_gemm_bf16<<<dim3(MROWS / 128, ceilDiv(NCLS, 32)), blk128, 0, stream>>>(
        imgb, DMODEL, txtb, DMODEL, MROWS, NCLS, DMODEL,
        nullptr, 0, Sbuf, nullptr, nullptr, 0, nullptr, nullptr);

    // 3) VQ argmin + per-batch top-1 via masked softmax accumulation
    argmin_kernel<<<dim3(ceilDiv(MROWS, 8)), blk256, 0, stream>>>(Sbuf, c2, minidx, MROWS, NCLS);
    topidx_kernel<<<dim3(ceilDiv(BATCH, 8)), blk256, 0, stream>>>(Sbuf, minidx, nimg, ntxt, topidx, BATCH, NCLS);

    // 4) per-class MLP vf[n] = fc2(relu(fc1(o_proj(v_proj(text[n])))))  [400 classes]
    //    (MHA collapses: single k/v token -> attention weights are identically 1)
    const unsigned mt400 = ceilDiv(NCLS, 128);   // 4
    wmma_gemm_bf16<<<dim3(mt400, DMODEL / 32), blk128, 0, stream>>>(
        txtb, DMODEL, vwb, DMODEL, NCLS, DMODEL, DMODEL,
        v_b, 0, y1f, y1b, nullptr, 0, nullptr, nullptr);
    wmma_gemm_bf16<<<dim3(mt400, DMODEL / 32), blk128, 0, stream>>>(
        y1b, DMODEL, owb, DMODEL, NCLS, DMODEL, DMODEL,
        o_b, 0, y2f, y2b, nullptr, 0, nullptr, nullptr);
    wmma_gemm_bf16<<<dim3(mt400, HID / 32), blk128, 0, stream>>>(
        y2b, DMODEL, f1b, DMODEL, NCLS, HID, DMODEL,
        fc1_b, 1, hf, hb, nullptr, 0, nullptr, nullptr);
    wmma_gemm_bf16<<<dim3(mt400, DMODEL / 32), blk128, 0, stream>>>(
        hb, HID, f2b, HID, NCLS, DMODEL, HID,
        fc2_b, 0, vff, nullptr, nullptr, 0, nullptr, nullptr);

    // 5) x = image + vf[top_idx[b]]; bf16 + row norms
    xbuild_kernel<<<dim3(ceilDiv(MROWS, 8)), blk256, 0, stream>>>(img, vff, topidx, xb, xnorm, MROWS, DMODEL);

    // 6) logits[a,b,t] = <x_n, txt_n> : raw GEMM scaled by 1/(|x||c|), stored transposed
    wmma_gemm_bf16<<<dim3(MROWS / 128, ceilDiv(NCLS, 32)), blk128, 0, stream>>>(
        xb, DMODEL, txtb, DMODEL, MROWS, NCLS, DMODEL,
        nullptr, 0, nullptr, nullptr, out, MROWS, xnorm, ntxt);
}
